// FullAttentionEstimator_47236050322118
// MI455X (gfx1250) — matrix operand order repaired and checked by reading, not compiled, not run on hardware
//
#include <hip/hip_runtime.h>
#include <hip/hip_fp16.h>

#define HID 128
#define TQ 512
#define TK 512
#define NB 2
#define EPSV 1e-5f

typedef __attribute__((ext_vector_type(16))) _Float16 v16h;
typedef __attribute__((ext_vector_type(8)))  _Float16 v8h;
typedef __attribute__((ext_vector_type(8)))  float    v8f;

union V16 { v16h v; v8h h8[2]; _Float16 e[16]; };
union V8F { v8f v; float e[8]; };

// ---- workspace layout (bytes) ----
static constexpr size_t OFF_QP  = 0;                       // 1024*128 f32
static constexpr size_t OFF_KP  = OFF_QP  + 1024*128*4;
static constexpr size_t OFF_UQ  = OFF_KP  + 1024*128*4;
static constexpr size_t OFF_VQ  = OFF_UQ  + 1024*128*4;
static constexpr size_t OFF_UK  = OFF_VQ  + 1024*128*4;
static constexpr size_t OFF_VK  = OFF_UK  + 1024*128*4;
static constexpr size_t OFF_SQ  = OFF_VK  + 1024*128*4;    // 1024 f32 each
static constexpr size_t OFF_SQQ = OFF_SQ  + 1024*4;
static constexpr size_t OFF_SK  = OFF_SQQ + 1024*4;
static constexpr size_t OFF_SKK = OFF_SK  + 1024*4;
static constexpr size_t OFF_W1U = OFF_SKK + 1024*4;        // 128*128 f32 each
static constexpr size_t OFF_W1V = OFF_W1U + 128*128*4;
static constexpr size_t OFF_W2U = OFF_W1V + 128*128*4;
static constexpr size_t OFF_W2V = OFF_W2U + 128*128*4;
static constexpr size_t OFF_W3UT= OFF_W2V + 128*128*4;     // f16 [h][c]
static constexpr size_t OFF_W3VT= OFF_W3UT+ 128*128*2;
static constexpr size_t OFF_GU  = OFF_W3VT+ 128*128*2;     // 128 f32 each
static constexpr size_t OFF_BUc = OFF_GU  + 512;
static constexpr size_t OFF_GV  = OFF_BUc + 512;
static constexpr size_t OFF_BVc = OFF_GV  + 512;

// ---------------------------------------------------------------------------
// Kernel 0: fold ln_g into Wu/Wv; emit W1*,W2* (f32 [c][h]), W3*ᵀ (f16 [h][c]),
// and Gu[h]=Σ g_c Wu[c][h], Bu[h]=Σ b_c Wu[c][h]+bu[h] (same for v).
// grid(128)=h, block(384)=c
// ---------------------------------------------------------------------------
__global__ void prep_kernel(const float* __restrict__ Wu, const float* __restrict__ bu,
                            const float* __restrict__ Wv, const float* __restrict__ bv,
                            const float* __restrict__ g,  const float* __restrict__ bln,
                            char* __restrict__ ws) {
  int h = blockIdx.x, c = threadIdx.x;
  __shared__ float r0[512], r1[512], r2[512], r3[512];
  float wu = Wu[c*HID + h], wv = Wv[c*HID + h];
  float gc = g[c], bc = bln[c];
  float gwu = gc*wu, gwv = gc*wv;
  float* W1u = (float*)(ws+OFF_W1U); float* W1v = (float*)(ws+OFF_W1V);
  float* W2u = (float*)(ws+OFF_W2U); float* W2v = (float*)(ws+OFF_W2V);
  _Float16* W3ut = (_Float16*)(ws+OFF_W3UT);
  _Float16* W3vt = (_Float16*)(ws+OFF_W3VT);
  if (c < 128)      { W1u[c*HID+h] = gwu;           W1v[c*HID+h] = gwv; }
  else if (c < 256) { W2u[(c-128)*HID+h] = gwu;     W2v[(c-128)*HID+h] = gwv; }
  else              { W3ut[h*HID+(c-256)] = (_Float16)gwu;
                      W3vt[h*HID+(c-256)] = (_Float16)gwv; }
  r0[c] = gwu; r1[c] = gwv; r2[c] = bc*wu; r3[c] = bc*wv;
  if (c < 128) { r0[384+c]=0.f; r1[384+c]=0.f; r2[384+c]=0.f; r3[384+c]=0.f; }
  __syncthreads();
  for (int s = 256; s > 0; s >>= 1) {
    if (c < s) { r0[c]+=r0[c+s]; r1[c]+=r1[c+s]; r2[c]+=r2[c+s]; r3[c]+=r3[c+s]; }
    __syncthreads();
  }
  if (c == 0) {
    ((float*)(ws+OFF_GU ))[h] = r0[0];
    ((float*)(ws+OFF_GV ))[h] = r1[0];
    ((float*)(ws+OFF_BUc))[h] = r2[0] + bu[h];
    ((float*)(ws+OFF_BVc))[h] = r3[0] + bv[h];
  }
}

// ---------------------------------------------------------------------------
// Kernel 1: projections qp=Q@Wq, kp=K@Wk (fp32, LDS-tiled).
// 2048 rows total (1024 q + 1024 k), 16 rows per block -> grid(128), block(256)
// ---------------------------------------------------------------------------
__global__ __launch_bounds__(256)
void proj_kernel(const float* __restrict__ Q, const float* __restrict__ K,
                 const float* __restrict__ Wq, const float* __restrict__ Wk,
                 char* __restrict__ ws) {
  __shared__ float Wt[64*128];
  __shared__ float Xr[16][68];
  int t = threadIdx.x;
  int rbase = blockIdx.x * 16;
  bool isK = rbase >= 1024;
  int r0 = isK ? rbase - 1024 : rbase;
  const float* X = isK ? K : Q;
  const float* W = isK ? Wk : Wq;
  float* OUT = (float*)(ws + (isK ? OFF_KP : OFF_QP));
  float acc[8] = {0.f,0.f,0.f,0.f,0.f,0.f,0.f,0.f};
  int h = t & 127, rg = (t >> 7) * 8;
  for (int kc = 0; kc < 16; ++kc) {
    for (int u = 0; u < 32; ++u) { int idx = t + 256*u; Wt[idx] = W[kc*64*128 + idx]; }
    for (int u = 0; u < 4; ++u) {
      int idx = t + 256*u; int rr = idx >> 6, cc = idx & 63;
      Xr[rr][cc] = X[(size_t)(r0+rr)*1024 + kc*64 + cc];
    }
    __syncthreads();
    for (int cc = 0; cc < 64; ++cc) {
      float wv = Wt[cc*128 + h];
      #pragma unroll
      for (int rr = 0; rr < 8; ++rr) acc[rr] += Xr[rg+rr][cc] * wv;
    }
    __syncthreads();
  }
  #pragma unroll
  for (int rr = 0; rr < 8; ++rr) OUT[(size_t)(r0+rg+rr)*HID + h] = acc[rr];
}

// ---------------------------------------------------------------------------
// Kernel 2: per-row sums (S, S²) and row projections Uq=qp@W1u etc.
// grid(2048) one row/block, block(128)
// ---------------------------------------------------------------------------
__global__ void rowstats_kernel(char* __restrict__ ws) {
  int r = blockIdx.x; bool isK = r >= 1024; int r0 = isK ? r-1024 : r;
  const float* P   = (const float*)(ws + (isK?OFF_KP :OFF_QP )) + (size_t)r0*HID;
  const float* Wu_ = (const float*)(ws + (isK?OFF_W2U:OFF_W1U));
  const float* Wv_ = (const float*)(ws + (isK?OFF_W2V:OFF_W1V));
  float* Uo = (float*)(ws + (isK?OFF_UK :OFF_UQ )) + (size_t)r0*HID;
  float* Vo = (float*)(ws + (isK?OFF_VK :OFF_VQ )) + (size_t)r0*HID;
  float* S  = (float*)(ws + (isK?OFF_SK :OFF_SQ ));
  float* SS = (float*)(ws + (isK?OFF_SKK:OFF_SQQ));
  __shared__ float row[128]; __shared__ float red[128];
  int h = threadIdx.x;
  row[h] = P[h]; __syncthreads();
  float au = 0.f, av = 0.f;
  for (int c = 0; c < 128; ++c) { float x = row[c]; au += x*Wu_[c*HID+h]; av += x*Wv_[c*HID+h]; }
  Uo[h] = au; Vo[h] = av;
  red[h] = row[h]; __syncthreads();
  for (int s = 64; s > 0; s >>= 1) { if (h < s) red[h] += red[h+s]; __syncthreads(); }
  if (h == 0) S[r0] = red[0];
  __syncthreads();
  red[h] = row[h]*row[h]; __syncthreads();
  for (int s = 64; s > 0; s >>= 1) { if (h < s) red[h] += red[h+s]; __syncthreads(); }
  if (h == 0) SS[r0] = red[0];
}

// ---------------------------------------------------------------------------
// Kernel 3: main pair kernel. 16i x 16j tile per workgroup, 8 waves,
// wave w owns h-block [16w,16w+16). W3 B-operands in VGPRs, per-i A-tile in LDS.
// grid(32,32,2), block(256)
// ---------------------------------------------------------------------------
__global__ __launch_bounds__(256)
void pair_kernel(const float* __restrict__ Wo, const float* __restrict__ bo,
                 const char* __restrict__ ws, float* __restrict__ out) {
  const int jb = blockIdx.x, ib = blockIdx.y, b = blockIdx.z;
  const int t = threadIdx.x, lane = t & 31, w = t >> 5;

  __shared__ float q32[16][132];
  __shared__ float k32[16][132];
  __shared__ __attribute__((aligned(16))) _Float16 A16[16][136];
  __shared__ float Uq_s[16][128], Vq_s[16][128], Uk_s[16][128], Vk_s[16][128];
  __shared__ float Sq_s[16], Sqq_s[16], Sk_s[16], Skk_s[16];
  __shared__ float mu_s[16][16], rs_s[16][16];
  __shared__ float Gu_s[128], Bu_s[128], Gv_s[128], Bv_s[128], Wo_s[128];
  __shared__ float part[8][16];

  const float* qp  = (const float*)(ws+OFF_QP);
  const float* kp  = (const float*)(ws+OFF_KP);
  const float* Uqg = (const float*)(ws+OFF_UQ);
  const float* Vqg = (const float*)(ws+OFF_VQ);
  const float* Ukg = (const float*)(ws+OFF_UK);
  const float* Vkg = (const float*)(ws+OFF_VK);
  const float* Sqg = (const float*)(ws+OFF_SQ);
  const float* Sqqg= (const float*)(ws+OFF_SQQ);
  const float* Skg = (const float*)(ws+OFF_SK);
  const float* Skkg= (const float*)(ws+OFF_SKK);
  const int rq = b*TQ + ib*16;   // global q-row base (0..1023)
  const int rk = b*TK + jb*16;   // global k-row base

  {
    int i = t >> 4, c0 = (t & 15) * 8;
    #pragma unroll
    for (int u = 0; u < 8; ++u) {
      q32[i][c0+u]  = qp [(size_t)(rq+i)*HID + c0+u];
      k32[i][c0+u]  = kp [(size_t)(rk+i)*HID + c0+u];
      Uq_s[i][c0+u] = Uqg[(size_t)(rq+i)*HID + c0+u];
      Vq_s[i][c0+u] = Vqg[(size_t)(rq+i)*HID + c0+u];
      Uk_s[i][c0+u] = Ukg[(size_t)(rk+i)*HID + c0+u];
      Vk_s[i][c0+u] = Vkg[(size_t)(rk+i)*HID + c0+u];
    }
  }
  if (t < 16) { Sq_s[t]=Sqg[rq+t]; Sqq_s[t]=Sqqg[rq+t]; Sk_s[t]=Skg[rk+t]; Skk_s[t]=Skkg[rk+t]; }
  if (t < 128) {
    Gu_s[t]=((const float*)(ws+OFF_GU ))[t]; Bu_s[t]=((const float*)(ws+OFF_BUc))[t];
    Gv_s[t]=((const float*)(ws+OFF_GV ))[t]; Bv_s[t]=((const float*)(ws+OFF_BVc))[t];
    Wo_s[t]=Wo[t];
  }

  // W3 B-operands into registers: lane L -> column h = hb + (L&15),
  // contiguous 16 K at base kc*32 + (L<16 ? 0 : 16)   (contiguous k order, same as A)
  const _Float16* w3u = (const _Float16*)(ws+OFF_W3UT);
  const _Float16* w3v = (const _Float16*)(ws+OFF_W3VT);
  const int hb = w * 16;
  const int hcol = hb + (lane & 15);
  V16 Bu_r[4], Bv_r[4];
  #pragma unroll
  for (int kc = 0; kc < 4; ++kc) {
    int cb = kc*32 + ((lane >> 4) << 4);
    Bu_r[kc].h8[0] = *(const v8h*)(w3u + (size_t)hcol*HID + cb);
    Bu_r[kc].h8[1] = *(const v8h*)(w3u + (size_t)hcol*HID + cb + 8);
    Bv_r[kc].h8[0] = *(const v8h*)(w3v + (size_t)hcol*HID + cb);
    Bv_r[kc].h8[1] = *(const v8h*)(w3v + (size_t)hcol*HID + cb + 8);
  }
  __syncthreads();

  // closed-form mean / rstd per pair: mu=(Sq+Sk+<q,k>)/384, E2=(Sqq+Skk+<q²,k²>)/384
  {
    int i = t >> 4, j = t & 15;
    float d1 = 0.f, d2 = 0.f;
    for (int c = 0; c < 128; ++c) {
      float p = q32[i][c] * k32[j][c];
      d1 += p; d2 += p*p;
    }
    float mu  = (Sq_s[i] + Sk_s[j] + d1) * (1.f/384.f);
    float e2  = (Sqq_s[i] + Skk_s[j] + d2) * (1.f/384.f);
    float var = e2 - mu*mu;
    mu_s[i][j] = mu;
    rs_s[i][j] = rsqrtf(var + EPSV);
  }
  __syncthreads();

  const float bo0 = bo[0];
  for (int i = 0; i < 16; ++i) {
    // build A tile: A[j][c] = f16( q_i[c] * k_j[c] )  (product formed in fp32)
    {
      int j = t & 15, c0 = (t >> 4) * 8;
      #pragma unroll
      for (int u = 0; u < 8; ++u)
        A16[j][c0+u] = (_Float16)(q32[i][c0+u] * k32[j][c0+u]);
    }
    __syncthreads();

    // 8 WMMAs: D(16j x 16h) += A(16j x 32c) * W3(32c x 16h), K=128 over 4 chunks
    v8f ua = {}; v8f va = {};
    const int jrow = lane & 15;
    const int kb = (lane >> 4) << 4;     // contiguous A-layout: 0 or 16
    #pragma unroll
    for (int kc = 0; kc < 4; ++kc) {
      V16 a;
      const _Float16* ap = &A16[jrow][kc*32 + kb];
      a.h8[0] = *(const v8h*)ap;
      a.h8[1] = *(const v8h*)(ap + 8);
      ua = __builtin_amdgcn_wmma_f32_16x16x32_f16(false, a.v, false, Bu_r[kc].v,
                                                  (short)0, ua, false, false);
      va = __builtin_amdgcn_wmma_f32_16x16x32_f16(false, a.v, false, Bv_r[kc].v,
                                                  (short)0, va, false, false);
    }

    // epilogue: affine layernorm terms, exact GELU, Wo dot, h-reduction
    V8F uu; uu.v = ua; V8F vv; vv.v = va;
    const int hh = hb + (lane & 15);
    const float Gu_h = Gu_s[hh], Bu_h = Bu_s[hh];
    const float Gv_h = Gv_s[hh], Bv_h = Bv_s[hh];
    const float Wo_h = Wo_s[hh];
    const float Uq_i = Uq_s[i][hh], Vq_i = Vq_s[i][hh];
    #pragma unroll
    for (int g = 0; g < 8; ++g) {
      int j = g + ((lane >> 4) << 3);    // ISA C/D layout: M = vgpr + 8*(lane/16)
      float mu = mu_s[i][j], rs = rs_s[i][j];
      float U = rs*(Uq_i + Uk_s[j][hh] + uu.e[g]) - rs*mu*Gu_h + Bu_h;
      float V = rs*(Vq_i + Vk_s[j][hh] + vv.e[g]) - rs*mu*Gv_h + Bv_h;
      float ge = 0.5f * V * (1.f + erff(V * 0.70710678118654752f));
      float val = U * ge * Wo_h;
      val += __shfl_xor(val, 1);
      val += __shfl_xor(val, 2);
      val += __shfl_xor(val, 4);
      val += __shfl_xor(val, 8);
      if ((lane & 15) == 0) part[w][j] = val;
    }
    __syncthreads();
    if (t < 16) {
      float o = bo0;
      #pragma unroll
      for (int ww = 0; ww < 8; ++ww) o += part[ww][t];
      out[((size_t)b*TQ + ib*16 + i) * TK + jb*16 + t] = o;
    }
    __syncthreads();
  }
}

// ---------------------------------------------------------------------------
extern "C" void kernel_launch(void* const* d_in, const int* in_sizes, int n_in,
                              void* d_out, int out_size, void* d_ws, size_t ws_size,
                              hipStream_t stream) {
  (void)in_sizes; (void)n_in; (void)out_size; (void)ws_size;
  const float* Q    = (const float*)d_in[0];
  const float* K    = (const float*)d_in[1];
  const float* Wq   = (const float*)d_in[2];
  const float* Wk   = (const float*)d_in[3];
  const float* ln_g = (const float*)d_in[4];
  const float* ln_b = (const float*)d_in[5];
  const float* Wu   = (const float*)d_in[6];
  const float* bu   = (const float*)d_in[7];
  const float* Wv   = (const float*)d_in[8];
  const float* bv   = (const float*)d_in[9];
  const float* Wo   = (const float*)d_in[10];
  const float* bo   = (const float*)d_in[11];
  char* ws = (char*)d_ws;

  prep_kernel<<<128, 384, 0, stream>>>(Wu, bu, Wv, bv, ln_g, ln_b, ws);
  proj_kernel<<<128, 256, 0, stream>>>(Q, K, Wq, Wk, ws);
  rowstats_kernel<<<2048, 128, 0, stream>>>(ws);
  dim3 grid(TK/16, TQ/16, NB);
  pair_kernel<<<grid, 256, 0, stream>>>(Wo, bo, (const char*)ws, (float*)d_out);
}
